// WindowAttention_3324304687550
// MI455X (gfx1250) — matrix operand order repaired and checked by reading, not compiled, not run on hardware
//
#include <hip/hip_runtime.h>

typedef __attribute__((ext_vector_type(16))) _Float16 v16h;
typedef __attribute__((ext_vector_type(8)))  _Float16 v8h;
typedef __attribute__((ext_vector_type(8)))  float    v8f;

#define BW_   4096
#define NTOK  64
#define DIM_  192
#define HEADS 6
#define DHEAD 32
#define NW_   64

// ---- workspace layout (bytes) ----
#define RPB_WS_OFF 0          // (6,64,64) f32   = 98304 B
#define RM_WS_OFF  98304      // (64,6,64,64) f32 = 6291456 B  (rpb + mask fused)

// ---- dynamic LDS partition (bytes) ----
#define XS_OFF   0            // 64 x 200 halfs : x tile, later O tile
#define QK_OFF   25600        // 64 x 392 halfs : cols 0..191 = qn, 192..383 = kn
#define VT_OFF   75776        // 6 x 32 x 72 halfs : v transposed per head
#define S_OFF    103424       // 2 x (64 x 65) f32 : scores; reused as weight pane Bs
#define P_OFF    136704       // 2 x (64 x 72) halfs : softmax probs
#define SMEM_BYTES 155136

#define XS_STR 200
#define QK_STR 392
#define VT_STR 72
#define S_STR  65
#define P_STR  72

static __device__ __forceinline__ v16h ld16(const _Float16* p0, const _Float16* p1) {
    v8h lo = *(const v8h*)p0;
    v8h hi = *(const v8h*)p1;
    return __builtin_shufflevector(lo, hi, 0,1,2,3,4,5,6,7,8,9,10,11,12,13,14,15);
}

// ---------------------------------------------------------------------------
// Kernel 1: continuous-position-bias MLP + gather -> rpb (6,64,64) f32
// ---------------------------------------------------------------------------
__global__ __launch_bounds__(256) void cpb_kernel(
    const float* __restrict__ table,   // (225,2)
    const float* __restrict__ w1,      // (512,2)
    const float* __restrict__ b1,      // (512)
    const float* __restrict__ w2,      // (6,512)
    const int*   __restrict__ rel_idx, // (64,64)
    float* __restrict__ rpb_ws)        // (6,64,64)
{
    __shared__ float tbl[225 * 6];
    for (int o = threadIdx.x; o < 225 * 6; o += 256) {
        int p = o / 6, hd = o % 6;
        float c0 = table[p * 2 + 0];
        float c1 = table[p * 2 + 1];
        float acc = 0.f;
        for (int t = 0; t < 512; ++t) {
            float hid = fmaxf(w1[t * 2] * c0 + w1[t * 2 + 1] * c1 + b1[t], 0.f);
            acc += hid * w2[hd * 512 + t];
        }
        tbl[o] = acc;
    }
    __syncthreads();
    for (int o = threadIdx.x; o < HEADS * NTOK * NTOK; o += 256) {
        int hd = o / (NTOK * NTOK), ij = o % (NTOK * NTOK);
        float v = tbl[rel_idx[ij] * 6 + hd];
        rpb_ws[o] = 16.0f / (1.0f + __expf(-v));
    }
}

// ---------------------------------------------------------------------------
// Kernel 2: fuse rpb + mask -> rm (NW, H, 64, 64): one load per S element later
// ---------------------------------------------------------------------------
__global__ __launch_bounds__(256) void rm_kernel(
    const float* __restrict__ rpb_ws,  // (6,64,64)
    const float* __restrict__ mask,    // (64,64,64)
    float* __restrict__ rm_ws)         // (64,6,64,64)
{
    int idx = blockIdx.x * 256 + threadIdx.x;      // 64*6*4096 total
    int ij  = idx & 4095;
    int wi  = idx / (HEADS * 4096);
    rm_ws[idx] = rpb_ws[idx % (HEADS * 4096)] + mask[wi * 4096 + ij];
}

// ---------------------------------------------------------------------------
// Kernel 3: fully fused window attention. One block per window (4096 blocks).
// ---------------------------------------------------------------------------
__global__ __launch_bounds__(256) void fused_attn(
    const float* __restrict__ x,           // (262144,192)
    const float* __restrict__ qkv_w,       // (576,192)
    const float* __restrict__ q_bias,      // (192)
    const float* __restrict__ v_bias,      // (192)
    const float* __restrict__ logit_scale, // (6)
    const float* __restrict__ proj_w,      // (192,192)
    const float* __restrict__ proj_b,      // (192)
    const float* __restrict__ rm_ws,       // (64,6,64,64)
    float* __restrict__ out)               // (262144,192)
{
    extern __shared__ unsigned char smem[];
    _Float16* xs  = (_Float16*)(smem + XS_OFF);
    _Float16* qk  = (_Float16*)(smem + QK_OFF);
    _Float16* vts = (_Float16*)(smem + VT_OFF);
    float*    S   = (float*)   (smem + S_OFF);
    _Float16* Bs  = (_Float16*)(smem + S_OFF);     // overlay
    _Float16* P   = (_Float16*)(smem + P_OFF);

    const int tid = threadIdx.x;
    const int b   = blockIdx.x;
    const int lane = tid & 31, wv = tid >> 5;
    const int lm = lane & 15, lh = lane >> 4;

    // ---- stage 1: load x tile (64x192 fp32 -> f16) -------------------------
    {
        const float* xg = x + (long)b * NTOK * DIM_;
        for (int it = 0; it < 12; ++it) {
            int e4  = tid + it * 256;
            int row = e4 / 48;
            int col = (e4 % 48) * 4;
            float4 f = *(const float4*)(xg + row * DIM_ + col);
            _Float16* dst = &xs[row * XS_STR + col];
            dst[0] = (_Float16)f.x; dst[1] = (_Float16)f.y;
            dst[2] = (_Float16)f.z; dst[3] = (_Float16)f.w;
        }
    }

    // ---- stage 2: QKV GEMM, 9 column blocks of 64 --------------------------
    for (int cb = 0; cb < 9; ++cb) {
        const int s        = cb / 3;               // uniform: 0=q, 1=k, 2=v
        const int rem_base = cb * 64 - s * DIM_;   // column within the 192 pane
        __syncthreads();
        for (int it = 0; it < 12; ++it) {
            int e4  = tid + it * 256;
            int n   = e4 / 48;
            int col = (e4 % 48) * 4;
            float4 f = *(const float4*)(qkv_w + (long)(cb * 64 + n) * DIM_ + col);
            _Float16* dst = &Bs[n * XS_STR + col];
            dst[0] = (_Float16)f.x; dst[1] = (_Float16)f.y;
            dst[2] = (_Float16)f.z; dst[3] = (_Float16)f.w;
        }
        __syncthreads();
        for (int ti = 0; ti < 2; ++ti) {
            int t  = wv * 2 + ti;                  // 16 tiles of the 64x64 block
            int mt = t >> 2, nt = t & 3;
            v8f acc = {};
            for (int kb = 0; kb < DIM_; kb += 32) {
                const _Float16* ar = &xs[(mt * 16 + lm) * XS_STR + kb];
                v16h a = ld16(ar + lh * 8, ar + 16 + lh * 8);
                const _Float16* br = &Bs[(nt * 16 + lm) * XS_STR + kb];
                v16h bb = ld16(br + lh * 8, br + 16 + lh * 8);
                acc = __builtin_amdgcn_wmma_f32_16x16x32_f16(
                    false, a, false, bb, (short)0, acc, false, false);
            }
            const int rem  = rem_base + nt * 16 + lm;
            const int rbase = mt * 16 + 8 * lh;
            if (s == 0) {
                const float bv = q_bias[rem];
                _Float16* dst = &qk[rbase * QK_STR + rem];
                for (int r = 0; r < 8; ++r) dst[r * QK_STR] = (_Float16)(acc[r] + bv);
            } else if (s == 1) {
                _Float16* dst = &qk[rbase * QK_STR + DIM_ + rem];
                for (int r = 0; r < 8; ++r) dst[r * QK_STR] = (_Float16)acc[r];
            } else {
                const float bv = v_bias[rem];
                const int hd = rem >> 5, j = rem & 31;
                _Float16* dst = &vts[hd * (DHEAD * VT_STR) + j * VT_STR + rbase];
                for (int r = 0; r < 8; ++r) dst[r] = (_Float16)(acc[r] + bv);
            }
        }
    }
    __syncthreads();

    // ---- stage 3: L2-normalize q and k rows per head -----------------------
    for (int o = tid; o < 2 * NTOK * HEADS; o += 256) {
        int which = o / (NTOK * HEADS);            // 0 = q, 1 = k
        int idx   = o % (NTOK * HEADS);
        int row   = idx / HEADS, hd = idx % HEADS;
        _Float16* p = &qk[row * QK_STR + which * DIM_ + hd * DHEAD];
        float ss = 0.f;
        for (int c = 0; c < DHEAD; ++c) { float v = (float)p[c]; ss += v * v; }
        float sc = 1.0f / fmaxf(sqrtf(ss), 1e-12f);
        for (int c = 0; c < DHEAD; ++c) p[c] = (_Float16)((float)p[c] * sc);
    }
    __syncthreads();

    // ---- stage 4: attention, 2 heads at a time (4 waves per head) ----------
    const int g  = wv >> 2;                        // head group 0/1
    const int wg = wv & 3;                         // wave within group
    const int wi = b & (NW_ - 1);
    for (int hp = 0; hp < 3; ++hp) {
        const int h = hp * 2 + g;
        const float lsc = __expf(fminf(logit_scale[h], 4.6051702f));  // min(ls, ln 100)
        const float* rmh = rm_ws + ((long)(wi * HEADS + h) << 12);    // (64,64) pane
        float*    Sg = S + g * (NTOK * S_STR);
        _Float16* Pg = P + g * (NTOK * P_STR);
        // S = qn @ kn^T  (K=32, single WMMA per 16x16 tile); mt = wg
        {
            const _Float16* ar = &qk[(wg * 16 + lm) * QK_STR + h * DHEAD];
            v16h a = ld16(ar + lh * 8, ar + 16 + lh * 8);
            for (int nt = 0; nt < 4; ++nt) {
                const _Float16* br = &qk[(nt * 16 + lm) * QK_STR + DIM_ + h * DHEAD];
                v16h bb = ld16(br + lh * 8, br + 16 + lh * 8);
                v8f c = {};
                c = __builtin_amdgcn_wmma_f32_16x16x32_f16(
                    false, a, false, bb, (short)0, c, false, false);
                const int gj = nt * 16 + lm;
                for (int r = 0; r < 8; ++r) {
                    int gi = wg * 16 + r + 8 * lh;
                    Sg[gi * S_STR + gj] = c[r] * lsc + rmh[gi * NTOK + gj];
                }
            }
        }
        __syncthreads();
        // softmax over rows (threads 0..63 of each 128-thread head group)
        {
            int tr = tid & 127;
            if (tr < NTOK) {
                float mx = -1e30f;
                for (int j = 0; j < NTOK; ++j) mx = fmaxf(mx, Sg[tr * S_STR + j]);
                float sum = 0.f;
                for (int j = 0; j < NTOK; ++j) {
                    float ev = __expf(Sg[tr * S_STR + j] - mx);
                    sum += ev;
                    Sg[tr * S_STR + j] = ev;
                }
                float inv = 1.0f / sum;
                for (int j = 0; j < NTOK; ++j)
                    Pg[tr * P_STR + j] = (_Float16)(Sg[tr * S_STR + j] * inv);
            }
        }
        __syncthreads();
        // O = P @ V  (K=64 -> 2 WMMAs); write into os (= dead xs pane)
        for (int nt = 0; nt < 2; ++nt) {
            v8f c = {};
            for (int kk = 0; kk < 2; ++kk) {
                const _Float16* ar = &Pg[(wg * 16 + lm) * P_STR + kk * 32];
                v16h a = ld16(ar + lh * 8, ar + 16 + lh * 8);
                const _Float16* br = &vts[h * (DHEAD * VT_STR) + (nt * 16 + lm) * VT_STR
                                          + kk * 32];
                v16h bb = ld16(br + lh * 8, br + 16 + lh * 8);
                c = __builtin_amdgcn_wmma_f32_16x16x32_f16(
                    false, a, false, bb, (short)0, c, false, false);
            }
            const int col = h * DHEAD + nt * 16 + lm;
            _Float16* dst = &xs[(wg * 16 + 8 * lh) * XS_STR + col];
            for (int r = 0; r < 8; ++r) dst[r * XS_STR] = (_Float16)c[r];
        }
        // next-iteration barriers order the P-pane reuse; none needed here
    }
    __syncthreads();

    // ---- stage 5: output projection, 3 column blocks of 64 -----------------
    for (int cb = 0; cb < 3; ++cb) {
        __syncthreads();
        for (int it = 0; it < 12; ++it) {
            int e4  = tid + it * 256;
            int n   = e4 / 48;
            int col = (e4 % 48) * 4;
            float4 f = *(const float4*)(proj_w + (long)(cb * 64 + n) * DIM_ + col);
            _Float16* dst = &Bs[n * XS_STR + col];
            dst[0] = (_Float16)f.x; dst[1] = (_Float16)f.y;
            dst[2] = (_Float16)f.z; dst[3] = (_Float16)f.w;
        }
        __syncthreads();
        for (int ti = 0; ti < 2; ++ti) {
            int t  = wv * 2 + ti;
            int mt = t >> 2, nt = t & 3;
            v8f acc = {};
            for (int kb = 0; kb < DIM_; kb += 32) {
                const _Float16* ar = &xs[(mt * 16 + lm) * XS_STR + kb];
                v16h a = ld16(ar + lh * 8, ar + 16 + lh * 8);
                const _Float16* br = &Bs[(nt * 16 + lm) * XS_STR + kb];
                v16h bb = ld16(br + lh * 8, br + 16 + lh * 8);
                acc = __builtin_amdgcn_wmma_f32_16x16x32_f16(
                    false, a, false, bb, (short)0, acc, false, false);
            }
            const int col = cb * 64 + nt * 16 + lm;
            const float bv = proj_b[col];
            float* dst = &out[((long)b * NTOK + mt * 16 + 8 * lh) * DIM_ + col];
            for (int r = 0; r < 8; ++r) dst[r * DIM_] = acc[r] + bv;
        }
    }
}

// ---------------------------------------------------------------------------
extern "C" void kernel_launch(void* const* d_in, const int* in_sizes, int n_in,
                              void* d_out, int out_size, void* d_ws, size_t ws_size,
                              hipStream_t stream) {
    const float* x           = (const float*)d_in[0];
    const float* mask        = (const float*)d_in[1];
    const float* qkv_w       = (const float*)d_in[2];
    const float* q_bias      = (const float*)d_in[3];
    const float* v_bias      = (const float*)d_in[4];
    const float* logit_scale = (const float*)d_in[5];
    const float* cpb_w1      = (const float*)d_in[6];
    const float* cpb_b1      = (const float*)d_in[7];
    const float* cpb_w2      = (const float*)d_in[8];
    const float* proj_w      = (const float*)d_in[9];
    const float* proj_b      = (const float*)d_in[10];
    const float* rct         = (const float*)d_in[11];
    const int*   rpi         = (const int*)d_in[12];
    float*       out         = (float*)d_out;
    float*       rpb_ws      = (float*)((char*)d_ws + RPB_WS_OFF);
    float*       rm_ws       = (float*)((char*)d_ws + RM_WS_OFF);

    (void)hipFuncSetAttribute((const void*)fused_attn,
                              hipFuncAttributeMaxDynamicSharedMemorySize, SMEM_BYTES);

    cpb_kernel<<<dim3(1), dim3(256), 0, stream>>>(rct, cpb_w1, cpb_b1, cpb_w2, rpi, rpb_ws);
    rm_kernel<<<dim3(NW_ * HEADS * NTOK * NTOK / 256), dim3(256), 0, stream>>>(
        rpb_ws, mask, rm_ws);
    fused_attn<<<dim3(BW_), dim3(256), SMEM_BYTES, stream>>>(
        x, qkv_w, q_bias, v_bias, logit_scale, proj_w, proj_b, rm_ws, out);
}
